// PositLinear_3212635538024
// MI455X (gfx1250) — matrix operand order repaired and checked by reading, not compiled, not run on hardware
//
#include <hip/hip_runtime.h>

// ---- CDNA5 (gfx1250) wave32 WMMA types ----
typedef __bf16 v16bf  __attribute__((ext_vector_type(16)));
typedef __bf16 bf16x8 __attribute__((ext_vector_type(8)));
typedef __bf16 bf16x4 __attribute__((ext_vector_type(4)));
typedef float  v8f    __attribute__((ext_vector_type(8)));

// Problem constants (from reference)
constexpr int TOKENS   = 8192;
constexpr int SIZE_IN  = 4096;   // K
constexpr int SIZE_OUT = 4096;   // N

// Tiling
constexpr int BM = 128, BN = 128, BK = 32;
constexpr int LDSP = 40;          // padded LDS stride (bf16 elems): 80B rows, 16B-aligned

// Split fp32 -> bf16 hi (truncate) + bf16 lo (residual, truncate)
__device__ __forceinline__ void split2(float f, __bf16& hi, __bf16& lo) {
    unsigned u  = __float_as_uint(f);
    unsigned hb = u & 0xffff0000u;
    union { unsigned short s; __bf16 b; } ch, cl;
    ch.s = (unsigned short)(hb >> 16);
    float r = f - __uint_as_float(hb);
    cl.s = (unsigned short)(__float_as_uint(r) >> 16);
    hi = ch.b; lo = cl.b;
}

// Build a 16-elem bf16 fragment from two contiguous 16B LDS chunks (-> 2x ds_load_b128)
__device__ __forceinline__ v16bf frag16(const __bf16* p0, const __bf16* p1) {
    bf16x8 a = *(const bf16x8*)p0;
    bf16x8 b = *(const bf16x8*)p1;
    return __builtin_shufflevector(a, b, 0,1,2,3,4,5,6,7,8,9,10,11,12,13,14,15);
}

// Coalesced global loads of a K-tile into registers (software pipeline stage)
__device__ __forceinline__ void gload(const float* __restrict__ x,
                                      const float* __restrict__ w,
                                      int bm, int bn, int kt, int tid,
                                      float4 ra[4], float4 rb[4]) {
#pragma unroll
    for (int i = 0; i < 4; ++i) {
        int f = tid + i * 256;
        int arow = f >> 3, ac4 = f & 7;           // A tile: 128 rows x 8 float4
        ra[i] = *(const float4*)(x + (size_t)(bm + arow) * SIZE_IN + kt + ac4 * 4);
        int brow = f >> 5, bc4 = f & 31;          // W tile: 32 rows x 32 float4
        rb[i] = *(const float4*)(w + (size_t)(kt + brow) * SIZE_OUT + bn + bc4 * 4);
    }
}

// Split to bf16 hi/lo and store to LDS (A row-major MxK; W transposed to NxK)
__device__ __forceinline__ void lstore(__bf16* sAh, __bf16* sAl,
                                       __bf16* sBh, __bf16* sBl,
                                       int tid, const float4 ra[4], const float4 rb[4]) {
#pragma unroll
    for (int i = 0; i < 4; ++i) {
        int f = tid + i * 256;
        int arow = f >> 3, ac = (f & 7) * 4;
        float av[4] = { ra[i].x, ra[i].y, ra[i].z, ra[i].w };
        bf16x4 h4, l4;
#pragma unroll
        for (int e = 0; e < 4; ++e) { __bf16 h, l; split2(av[e], h, l); h4[e] = h; l4[e] = l; }
        *(bf16x4*)(sAh + arow * LDSP + ac) = h4;   // 8B-aligned ds_store_b64
        *(bf16x4*)(sAl + arow * LDSP + ac) = l4;

        int brow = f >> 5, bc = (f & 31) * 4;
        float bv[4] = { rb[i].x, rb[i].y, rb[i].z, rb[i].w };
#pragma unroll
        for (int e = 0; e < 4; ++e) {
            __bf16 h, l; split2(bv[e], h, l);
            sBh[(bc + e) * LDSP + brow] = h;       // transpose: LDS holds W^T (N x K)
            sBl[(bc + e) * LDSP + brow] = l;
        }
    }
}

__global__ __launch_bounds__(256)
void positlinear_wmma_bf16x3(const float* __restrict__ x,
                             const float* __restrict__ w,
                             const float* __restrict__ bias,
                             float* __restrict__ out) {
    // Double-buffered LDS: 2 x 40 KB of the 320 KB WGP pool
    __shared__ __bf16 sAh[2][BM * LDSP];
    __shared__ __bf16 sAl[2][BM * LDSP];
    __shared__ __bf16 sBh[2][BN * LDSP];
    __shared__ __bf16 sBl[2][BN * LDSP];

    const int tid  = threadIdx.x;
    const int lane = tid & 31;
    const int wave = tid >> 5;          // 8 waves
    const int wm   = wave >> 2;         // 0..1 -> 64 rows each
    const int wn   = wave & 3;          // 0..3 -> 32 cols each
    const int lh   = (lane >> 4) & 1;   // lane half
    const int l15  = lane & 15;

    const int bm = blockIdx.y * BM;
    const int bn = blockIdx.x * BN;

    float biasv[2];
#pragma unroll
    for (int nt = 0; nt < 2; ++nt)
        biasv[nt] = bias[bn + wn * 32 + nt * 16 + l15];

    v8f acc[4][2];
#pragma unroll
    for (int mt = 0; mt < 4; ++mt)
#pragma unroll
        for (int nt = 0; nt < 2; ++nt)
#pragma unroll
            for (int e = 0; e < 8; ++e) acc[mt][nt][e] = 0.0f;

    // 3-deep pipeline prologue: LDS holds tile 0, registers hold tile 1
    float4 ra[4], rb[4];
    gload(x, w, bm, bn, 0, tid, ra, rb);
    lstore(sAh[0], sAl[0], sBh[0], sBl[0], tid, ra, rb);
    gload(x, w, bm, bn, BK, tid, ra, rb);
    __syncthreads();

    int p = 0;
    for (int kt = 0; kt < SIZE_IN; kt += BK, p ^= 1) {
        // Stage tile kt+BK into the other LDS buffer while computing tile kt
        if (kt + BK < SIZE_IN)
            lstore(sAh[p ^ 1], sAl[p ^ 1], sBh[p ^ 1], sBl[p ^ 1], tid, ra, rb);
        // Fetch tile kt+2*BK into registers (latency hidden behind WMMAs)
        if (kt + 2 * BK < SIZE_IN)
            gload(x, w, bm, bn, kt + 2 * BK, tid, ra, rb);
        // Touch tile kt+3*BK toward L2 (gfx1250 global_prefetch_b8)
        if (kt + 3 * BK < SIZE_IN) {
            __builtin_prefetch(x + (size_t)(bm + (tid >> 3)) * SIZE_IN + kt + 3 * BK, 0, 0);
            __builtin_prefetch(w + (size_t)(kt + 3 * BK + (tid >> 5)) * SIZE_OUT + bn, 0, 0);
        }

        // B fragments (W^T rows are K-contiguous): v0-3 K=0..7(+8 hi-lanes), v4-7 K=16..23(+8 hi-lanes)
        v16bf bh[2], bl[2];
#pragma unroll
        for (int nt = 0; nt < 2; ++nt) {
            const __bf16* pbh = sBh[p] + (wn * 32 + nt * 16 + l15) * LDSP + (lh ? 8 : 0);
            const __bf16* pbl = sBl[p] + (wn * 32 + nt * 16 + l15) * LDSP + (lh ? 8 : 0);
            bh[nt] = frag16(pbh, pbh + 16);
            bl[nt] = frag16(pbl, pbl + 16);
        }

#pragma unroll
        for (int mt = 0; mt < 4; ++mt) {
            // A fragment: v0-3 K=0..7(+8 hi-lanes), v4-7 K=16..23(+8 hi-lanes)
            const __bf16* pah = sAh[p] + (wm * 64 + mt * 16 + l15) * LDSP + (lh ? 8 : 0);
            const __bf16* pal = sAl[p] + (wm * 64 + mt * 16 + l15) * LDSP + (lh ? 8 : 0);
            v16bf ah = frag16(pah, pah + 16);
            v16bf al = frag16(pal, pal + 16);
#pragma unroll
            for (int nt = 0; nt < 2; ++nt) {
                acc[mt][nt] = __builtin_amdgcn_wmma_f32_16x16x32_bf16(
                    false, ah, false, bh[nt], (short)0, acc[mt][nt], false, false);
                acc[mt][nt] = __builtin_amdgcn_wmma_f32_16x16x32_bf16(
                    false, ah, false, bl[nt], (short)0, acc[mt][nt], false, false);
                acc[mt][nt] = __builtin_amdgcn_wmma_f32_16x16x32_bf16(
                    false, al, false, bh[nt], (short)0, acc[mt][nt], false, false);
            }
        }

        // Single barrier per iteration: protects buffer p (read this iter, written
        // next iter) and buffer p^1 (written this iter, read next iter).
        __syncthreads();
    }

    // Epilogue: C layout = VGPR r -> M = r + 8*laneHi, N = lane%16
#pragma unroll
    for (int mt = 0; mt < 4; ++mt) {
#pragma unroll
        for (int nt = 0; nt < 2; ++nt) {
            const int col     = bn + wn * 32 + nt * 16 + l15;
            const int rowbase = bm + wm * 64 + mt * 16 + lh * 8;
#pragma unroll
            for (int r = 0; r < 8; ++r)
                out[(size_t)(rowbase + r) * SIZE_OUT + col] = acc[mt][nt][r] + biasv[nt];
        }
    }
}

extern "C" void kernel_launch(void* const* d_in, const int* in_sizes, int n_in,
                              void* d_out, int out_size, void* d_ws, size_t ws_size,
                              hipStream_t stream) {
    (void)in_sizes; (void)n_in; (void)out_size; (void)d_ws; (void)ws_size;
    const float* x    = (const float*)d_in[0];
    const float* w    = (const float*)d_in[1];
    const float* bias = (const float*)d_in[2];
    float*       out  = (float*)d_out;

    dim3 grid(SIZE_OUT / BN, TOKENS / BM);   // 32 x 64 = 2048 blocks
    positlinear_wmma_bf16x3<<<grid, 256, 0, stream>>>(x, w, bias, out);
}
